// TorchLMHeadCE_65472481460588
// MI455X (gfx1250) — matrix operand order repaired and checked by reading, not compiled, not run on hardware
//
#include <hip/hip_runtime.h>
#include <hip/hip_bf16.h>

// ---- problem constants (match reference) ----
#define N_TOK   4096
#define HDIM    4096
#define VOCAB   32000
#define IGNORE_INDEX (-100LL)

// ---- tiling ----
#define MTILE   64
#define NTILE   64
#define KSTEP   32
#define SPLITV  20                                   // V-chunks per M-tile row
#define NTILES_TOTAL   (VOCAB / NTILE)               // 500
#define TILES_PER_CHUNK (NTILES_TOTAL / SPLITV)      // 25
#define MTILES  (N_TOK / MTILE)                      // 64
#define NPART   (SPLITV * 2)                         // 2 column-halves per chunk

typedef __attribute__((ext_vector_type(16))) __bf16 v16bf;
typedef __attribute__((ext_vector_type(8)))  float  v8f;

union ABFrag {
  v16bf v;
  uint4 q[2];
};

// -----------------------------------------------------------------------------
// Kernel 1: fused GEMM (x @ W^T) + per-row online (max, sum-exp) statistics.
// Grid: (MTILES, SPLITV); block = 128 threads = 4 waves (wave32).
// Wave w: rows (w&1)*32..+31 of the 64-row tile, cols (w>>1)*32..+31.
// Each wave keeps running softmax stats for its 32 rows x 32-col half, written
// as partials; finalize kernel merges 2*SPLITV partials per row.
// -----------------------------------------------------------------------------
__global__ __launch_bounds__(128)
void lmhead_lse_kernel(const __bf16* __restrict__ X,
                       const __bf16* __restrict__ W,
                       float2* __restrict__ partials) {
  const int lane   = threadIdx.x & 31;
  const int wave   = threadIdx.x >> 5;
  const int lmod   = lane & 15;
  const int laneHi = (lane >= 16) ? 1 : 0;
  const int mtile  = blockIdx.x;
  const int chunk  = blockIdx.y;
  const int rowHalf = wave & 1;    // which 32 rows of the 64-row tile
  const int colHalf = wave >> 1;   // which 32 cols of the 64-col tile

  const int mBase = mtile * MTILE + rowHalf * 32;

  // A fragment addressing: lane holds row M = lane%16 (both lane halves hold
  // the same rows, different K); K offset 0/16 by lane half, +8 for 2nd qword.
  const __bf16* arow0 = X + (size_t)(mBase + lmod) * HDIM;        // rows +0..15
  const __bf16* arow1 = X + (size_t)(mBase + 16 + lmod) * HDIM;   // rows +16..31
  const int aoff = laneHi ? 16 : 0;
  const int boff = laneHi ? 16 : 0;  // B: lanes 0-15 K=0..15, 16-31 K=16..31

  // running stats per (im, r): rows 16*im + r + 8*laneHi (replicated across the
  // 16 lanes of each half after the shuffle reductions)
  float m_run[2][8], s_run[2][8];
#pragma unroll
  for (int im = 0; im < 2; ++im)
#pragma unroll
    for (int r = 0; r < 8; ++r) { m_run[im][r] = -__builtin_inff(); s_run[im][r] = 0.0f; }

  for (int t = 0; t < TILES_PER_CHUNK; ++t) {
    const int nBase = (chunk * TILES_PER_CHUNK + t) * NTILE + colHalf * 32;
    // B fragment: lane holds column N = lane%16 of the 32x16 block, i.e. a
    // contiguous run of K along row (nBase + lmod) of W (row-major) -> b128s.
    const __bf16* bcol0 = W + (size_t)(nBase + lmod) * HDIM;       // cols +0..15
    const __bf16* bcol1 = W + (size_t)(nBase + 16 + lmod) * HDIM;  // cols +16..31

    v8f acc00 = {}, acc01 = {}, acc10 = {}, acc11 = {};

    for (int k0 = 0; k0 < HDIM; k0 += KSTEP) {
      ABFrag a0, a1, b0, b1;
      a0.q[0] = *(const uint4*)(arow0 + k0 + aoff);
      a0.q[1] = *(const uint4*)(arow0 + k0 + aoff + 8);
      a1.q[0] = *(const uint4*)(arow1 + k0 + aoff);
      a1.q[1] = *(const uint4*)(arow1 + k0 + aoff + 8);
      b0.q[0] = *(const uint4*)(bcol0 + k0 + boff);
      b0.q[1] = *(const uint4*)(bcol0 + k0 + boff + 8);
      b1.q[0] = *(const uint4*)(bcol1 + k0 + boff);
      b1.q[1] = *(const uint4*)(bcol1 + k0 + boff + 8);

      // prefetch the streaming W operand a few K-steps ahead
      if (k0 + 4 * KSTEP < HDIM) {
        __builtin_prefetch(bcol0 + k0 + boff + 4 * KSTEP, 0, 3);
        __builtin_prefetch(bcol1 + k0 + boff + 4 * KSTEP, 0, 3);
      }

      acc00 = __builtin_amdgcn_wmma_f32_16x16x32_bf16(false, a0.v, false, b0.v,
                                                      (short)0, acc00, false, false);
      acc01 = __builtin_amdgcn_wmma_f32_16x16x32_bf16(false, a0.v, false, b1.v,
                                                      (short)0, acc01, false, false);
      acc10 = __builtin_amdgcn_wmma_f32_16x16x32_bf16(false, a1.v, false, b0.v,
                                                      (short)0, acc10, false, false);
      acc11 = __builtin_amdgcn_wmma_f32_16x16x32_bf16(false, a1.v, false, b1.v,
                                                      (short)0, acc11, false, false);
    }

    // Online softmax update over this 32(row) x 32(col) logits block.
    // C layout: VGPR r of a 16x16 acc holds row (r + 8*laneHi), col = lane%16.
#pragma unroll
    for (int im = 0; im < 2; ++im) {
      v8f c0 = im ? acc10 : acc00;
      v8f c1 = im ? acc11 : acc01;
#pragma unroll
      for (int r = 0; r < 8; ++r) {
        float v0 = c0[r];
        float v1 = c1[r];
        float mt = fmaxf(v0, v1);
#pragma unroll
        for (int msk = 1; msk <= 8; msk <<= 1)
          mt = fmaxf(mt, __shfl_xor(mt, msk, 32));   // reduce across 16-lane half
        float mold = m_run[im][r];
        float mnew = fmaxf(mold, mt);
        float p = __expf(v0 - mnew) + __expf(v1 - mnew);
#pragma unroll
        for (int msk = 1; msk <= 8; msk <<= 1)
          p += __shfl_xor(p, msk, 32);
        s_run[im][r] = s_run[im][r] * __expf(mold - mnew) + p;
        m_run[im][r] = mnew;
      }
    }
  }

  // spill per-row partial (m, s) for this column half
  if (lmod == 0) {
#pragma unroll
    for (int im = 0; im < 2; ++im)
#pragma unroll
      for (int r = 0; r < 8; ++r) {
        int row64 = rowHalf * 32 + 16 * im + r + 8 * laneHi;
        size_t idx = ((size_t)mtile * NPART + (chunk * 2 + colHalf)) * MTILE + row64;
        partials[idx] = make_float2(m_run[im][r], s_run[im][r]);
      }
  }
}

// -----------------------------------------------------------------------------
// Kernel 2: target logit t[i] = dot(x[i,:], W[y[i],:]) in fp32. One wave / row.
// -----------------------------------------------------------------------------
__global__ __launch_bounds__(128)
void lmhead_target_dot(const __bf16* __restrict__ X,
                       const __bf16* __restrict__ W,
                       const long long* __restrict__ Y,
                       float* __restrict__ tlogit) {
  const int gw   = (int)((blockIdx.x * blockDim.x + threadIdx.x) >> 5);
  const int lane = threadIdx.x & 31;
  if (gw >= N_TOK) return;

  long long yi = Y[gw];
  long long ys = (yi == IGNORE_INDEX) ? 0 : yi;
  const __bf16* xr = X + (size_t)gw * HDIM;
  const __bf16* wr = W + (size_t)ys * HDIM;

  float acc = 0.0f;
  for (int h = lane * 8; h < HDIM; h += 32 * 8) {
    union { uint4 q; __bf16 e[8]; } xa, wa;
    xa.q = *(const uint4*)(xr + h);
    wa.q = *(const uint4*)(wr + h);
#pragma unroll
    for (int j = 0; j < 8; ++j)
      acc += (float)xa.e[j] * (float)wa.e[j];
  }
#pragma unroll
  for (int msk = 1; msk < 32; msk <<= 1)
    acc += __shfl_xor(acc, msk, 32);
  if (lane == 0) tlogit[gw] = acc;
}

// -----------------------------------------------------------------------------
// Kernel 3: merge partials -> lse per row, nll = lse - t, masked mean -> out[0]
// -----------------------------------------------------------------------------
__global__ __launch_bounds__(256)
void lmhead_finalize(const float2* __restrict__ partials,
                     const float* __restrict__ tlogit,
                     const long long* __restrict__ Y,
                     float* __restrict__ out) {
  __shared__ float ssum[256];
  __shared__ int   scnt[256];

  float sum = 0.0f;
  int   cnt = 0;
  for (int row = threadIdx.x; row < N_TOK; row += 256) {
    const int mtile = row / MTILE;
    const int r     = row % MTILE;
    const float2* p = partials + (size_t)mtile * NPART * MTILE + r;

    float M = -__builtin_inff();
    for (int c = 0; c < NPART; ++c)
      M = fmaxf(M, p[(size_t)c * MTILE].x);
    float S = 0.0f;
    for (int c = 0; c < NPART; ++c) {
      float2 ms = p[(size_t)c * MTILE];
      S += ms.y * __expf(ms.x - M);
    }
    float lse = M + __logf(S);

    long long yi = Y[row];
    if (yi != IGNORE_INDEX) {
      sum += lse - tlogit[row];
      cnt += 1;
    }
  }

  ssum[threadIdx.x] = sum;
  scnt[threadIdx.x] = cnt;
  __syncthreads();
  for (int s = 128; s > 0; s >>= 1) {
    if ((int)threadIdx.x < s) {
      ssum[threadIdx.x] += ssum[threadIdx.x + s];
      scnt[threadIdx.x] += scnt[threadIdx.x + s];
    }
    __syncthreads();
  }
  if (threadIdx.x == 0) {
    int   c = scnt[0] > 1 ? scnt[0] : 1;
    out[0] = ssum[0] / (float)c;
  }
}

// -----------------------------------------------------------------------------
extern "C" void kernel_launch(void* const* d_in, const int* in_sizes, int n_in,
                              void* d_out, int out_size, void* d_ws, size_t ws_size,
                              hipStream_t stream) {
  const __bf16*     X = (const __bf16*)d_in[0];      // [N, H] bf16
  const __bf16*     W = (const __bf16*)d_in[1];      // [V, H] bf16
  const long long*  Y = (const long long*)d_in[2];   // [N] int64
  float*          out = (float*)d_out;               // scalar fp32

  float2* partials = (float2*)d_ws;                                  // MTILES*NPART*MTILE
  size_t  partBytes = (size_t)MTILES * NPART * MTILE * sizeof(float2);
  float*  tlogit   = (float*)((char*)d_ws + partBytes);              // N_TOK floats

  dim3 grid(MTILES, SPLITV);       // x fastest: all M-tiles share a W slice in L2
  lmhead_lse_kernel<<<grid, 128, 0, stream>>>(X, W, partials);
  lmhead_target_dot<<<(N_TOK * 32) / 128, 128, 0, stream>>>(X, W, Y, tlogit);
  lmhead_finalize<<<1, 256, 0, stream>>>(partials, tlogit, Y, out);
}
